// ConversationGCN_26637387170410
// MI455X (gfx1250) — matrix operand order repaired and checked by reading, not compiled, not run on hardware
//
#include <hip/hip_runtime.h>

// ---------------------------------------------------------------------------
// ConversationGCN on MI455X (gfx1250, wave32).
//  - All GEMMs via v_wmma_f32_16x16x32_bf16 (fp32 -> bf16 inputs, f32 accum).
//  - gfx1250 async global->LDS staging (ASYNCcnt) with manual fallback.
//  - LSTM recurrence: persistent blocks, Whh register-resident (no LDS bank
//    conflicts), h broadcast via ds_load_b128, next-step z prefetched.
//  - GCN aggregation: f32 global atomics + global_prefetch.
// ---------------------------------------------------------------------------

typedef __attribute__((ext_vector_type(16))) __bf16 bf16x16;
typedef __attribute__((ext_vector_type(8)))  float  f32x8;

// async-to-LDS builtin operand types: pointer to int4 (vector_size(16)),
// global (AS1) source and LDS (AS3) destination.
typedef int v4i_ __attribute__((vector_size(16)));
typedef __attribute__((address_space(1))) v4i_ as1_v4i;
typedef __attribute__((address_space(3))) v4i_ as3_v4i;

#define NN 30000
#define NE 960000

#if defined(__gfx1250__) && __has_builtin(__builtin_amdgcn_global_load_async_to_lds_b128)
#define USE_ASYNC_COPY 1
#else
#define USE_ASYNC_COPY 0
#endif

__device__ __forceinline__ void wait_async0() {
#if defined(__gfx1250__)
#if __has_builtin(__builtin_amdgcn_s_wait_asynccnt)
  __builtin_amdgcn_s_wait_asynccnt(0);
#else
  asm volatile("s_wait_asynccnt 0x0" ::: "memory");
#endif
#endif
}

// ---------------- elementwise helpers ----------------
__global__ void k_cast_bf16(const float* __restrict__ in, __bf16* __restrict__ out, int n) {
  int i = (blockIdx.x * blockDim.x + threadIdx.x) * 4;
  if (i + 3 < n) {
    float4 v = *(const float4*)(in + i);
    out[i+0] = (__bf16)v.x; out[i+1] = (__bf16)v.y;
    out[i+2] = (__bf16)v.z; out[i+3] = (__bf16)v.w;
  }
}

__global__ void k_zero_f32(float* __restrict__ p, int n) {
  int i = (blockIdx.x * blockDim.x + threadIdx.x) * 4;
  if (i + 3 < n) *(float4*)(p + i) = make_float4(0.f, 0.f, 0.f, 0.f);
}

// W[R,C] f32 -> out[C,R] bf16 (row-major [Nout,K] "Bt" layout for the GEMM)
__global__ void k_transpose_cast(const float* __restrict__ in, __bf16* __restrict__ out,
                                 int R, int C) {
  int i = blockIdx.x * blockDim.x + threadIdx.x;
  if (i < R * C) { int r = i / C, c = i - r * C; out[c * R + r] = (__bf16)in[i]; }
}

// ---------------- WMMA GEMM: C[M,N] = A[M,K] @ Bt[N,K]^T (+bias) ----------------
// Block = 128 threads = 4 waves. Block tile 64(M) x 64(N), K step 32.
// Wave w owns rows [w*16, w*16+16) and all 64 columns via 4 f32x8 accumulators.
__global__ __launch_bounds__(128)
void k_gemm_bf16(const __bf16* __restrict__ A, const __bf16* __restrict__ Bt,
                 const float* __restrict__ bias, float* __restrict__ C,
                 int M, int N, int K) {
  __shared__ __bf16 As[64 * 32];
  __shared__ __bf16 Bs[64 * 32];
  const int tid  = threadIdx.x;
  const int lane = tid & 31;
  const int wave = tid >> 5;
  const int hi   = lane >> 4;          // half-wave select
  const int lo   = lane & 15;
  const int mBase = blockIdx.x * 64;
  const int nBase = blockIdx.y * 64;

  f32x8 acc[4];
  #pragma unroll
  for (int j = 0; j < 4; ++j)
    #pragma unroll
    for (int r = 0; r < 8; ++r) acc[j][r] = 0.0f;

  const int loadRow = tid >> 1;          // 0..63
  const int loadCol = (tid & 1) * 16;    // 0 or 16 (bf16 elements)

  for (int k0 = 0; k0 < K; k0 += 32) {
    __syncthreads();                     // LDS reads of previous step done
    {
      const int gr = mBase + loadRow;
#if USE_ASYNC_COPY
      // gfx1250 async gather: each lane moves 2x16B of A and B straight into
      // LDS (tracked by ASYNCcnt). Rows >= M over-read inside the workspace
      // arena; their products land in accumulators whose stores are M-guarded.
      const __bf16* gA = A + (size_t)gr * K + k0 + loadCol;
      const __bf16* gB = Bt + (size_t)(nBase + loadRow) * K + k0 + loadCol;
      __bf16* lA = As + loadRow * 32 + loadCol;
      __bf16* lB = Bs + loadRow * 32 + loadCol;
      __builtin_amdgcn_global_load_async_to_lds_b128(
          (as1_v4i*)gA, (as3_v4i*)lA, 0, 0);
      __builtin_amdgcn_global_load_async_to_lds_b128(
          (as1_v4i*)(gA + 8), (as3_v4i*)(lA + 8), 0, 0);
      __builtin_amdgcn_global_load_async_to_lds_b128(
          (as1_v4i*)gB, (as3_v4i*)lB, 0, 0);
      __builtin_amdgcn_global_load_async_to_lds_b128(
          (as1_v4i*)(gB + 8), (as3_v4i*)(lB + 8), 0, 0);
      wait_async0();
#else
      uint4 a0 = make_uint4(0,0,0,0), a1 = make_uint4(0,0,0,0);
      if (gr < M) {
        const uint4* gp = (const uint4*)(A + (size_t)gr * K + k0 + loadCol);
        a0 = gp[0]; a1 = gp[1];
      }
      *(uint4*)(As + loadRow * 32 + loadCol)     = a0;
      *(uint4*)(As + loadRow * 32 + loadCol + 8) = a1;
      const uint4* bp = (const uint4*)(Bt + (size_t)(nBase + loadRow) * K + k0 + loadCol);
      *(uint4*)(Bs + loadRow * 32 + loadCol)     = bp[0];
      *(uint4*)(Bs + loadRow * 32 + loadCol + 8) = bp[1];
#endif
    }
    __syncthreads();                     // all waves' tiles resident

    // A fragment (16x32) in this code's contiguous k order, not the ISA table order:
    //  VGPR v : K = hi*16 + 2v,2v+1 (contiguous, matches B fragment)
    union { unsigned int u[8]; bf16x16 v; } afrag;
    const __bf16* arow = As + (wave * 16 + lo) * 32;
    #pragma unroll
    for (int vv = 0; vv < 8; ++vv) {
      int kb = hi * 16 + 2 * vv;
      afrag.u[vv] = *(const unsigned int*)(arow + kb);
    }
    #pragma unroll
    for (int j = 0; j < 4; ++j) {
      // B fragment in contiguous k order (same as A, not ISA table order): lanes0-15 K=0..15, lanes16-31 K=16..31
      union { unsigned int u[8]; bf16x16 v; } bfrag;
      const __bf16* brow = Bs + (j * 16 + lo) * 32 + hi * 16;
      #pragma unroll
      for (int vv = 0; vv < 8; ++vv)
        bfrag.u[vv] = *(const unsigned int*)(brow + 2 * vv);
      acc[j] = __builtin_amdgcn_wmma_f32_16x16x32_bf16(
          false, afrag.v, false, bfrag.v, (short)0, acc[j], false, false);
    }
  }

  // C/D layout: VGPR r, lanes0-15 -> M=r, lanes16-31 -> M=8+r; N = lane&15
  #pragma unroll
  for (int j = 0; j < 4; ++j) {
    int n = nBase + j * 16 + lo;
    float bv = bias ? bias[n] : 0.0f;
    #pragma unroll
    for (int r = 0; r < 8; ++r) {
      int m = mBase + wave * 16 + hi * 8 + r;
      if (m < M) C[(size_t)m * N + n] = acc[j][r] + bv;
    }
  }
}

// ---------------- LSTM recurrence (persistent, one block per direction) ----------------
// Z = x@Wih^T + b precomputed ([n,512] per direction). 512 threads: thread g
// keeps Whh[g,0:128] in VGPRs (no LDS bank conflicts); the h vector is the
// only shared state (conflict-free broadcast ds_load_b128). Next step's z is
// prefetched so its global-load latency overlaps the dot + barrier.
__global__ __launch_bounds__(512)
void k_lstm_layer(const float* __restrict__ zf, const float* __restrict__ zb,
                  const float* __restrict__ whhF, const float* __restrict__ whhB,
                  float* __restrict__ Hout, int n) {
  __shared__ float zsh[512];
  __shared__ float hsh[128];
  const int tid = threadIdx.x;
  const int dir = blockIdx.x;        // 0 = fwd, 1 = bwd
  const float* Z = dir ? zb : zf;
  const float* W = (dir ? whhB : whhF) + (size_t)tid * 128;

  float w[128];                      // register-resident Whh row (512B, 16B-aligned)
  #pragma unroll
  for (int k4 = 0; k4 < 32; ++k4) {
    float4 v = ((const float4*)W)[k4];
    w[4*k4+0] = v.x; w[4*k4+1] = v.y; w[4*k4+2] = v.z; w[4*k4+3] = v.w;
  }
  if (tid < 128) hsh[tid] = 0.0f;
  __syncthreads();

  float c = 0.0f;                    // cell state, owned by threads < 128
  int t = dir ? (n - 1) : 0;
  const int step = dir ? -1 : 1;
  float znext = Z[(size_t)t * 512 + tid];
  for (int s = 0; s < n; ++s, t += step) {
    float z = znext;
    if (s + 1 < n) znext = Z[(size_t)(t + step) * 512 + tid];  // overlapped prefetch
    float p0 = 0.f, p1 = 0.f, p2 = 0.f, p3 = 0.f;              // 4-way ILP
    #pragma unroll
    for (int k4 = 0; k4 < 32; ++k4) {
      float4 hv = *(const float4*)(hsh + 4 * k4);              // broadcast read
      p0 += w[4*k4+0] * hv.x;
      p1 += w[4*k4+1] * hv.y;
      p2 += w[4*k4+2] * hv.z;
      p3 += w[4*k4+3] * hv.w;
    }
    zsh[tid] = z + (p0 + p1) + (p2 + p3);
    __syncthreads();                 // all gate dots done; zsh complete
    if (tid < 128) {
      float zi = zsh[tid], zF = zsh[128 + tid], zg = zsh[256 + tid], zo = zsh[384 + tid];
      float ig = 1.0f / (1.0f + __expf(-zi));
      float fg = 1.0f / (1.0f + __expf(-zF));
      float og = 1.0f / (1.0f + __expf(-zo));
      c = fg * c + ig * tanhf(zg);
      float h = og * tanhf(c);
      hsh[tid] = h;
      Hout[(size_t)t * 256 + dir * 128 + tid] = h;
    }
    __syncthreads();                 // hsh visible for next step
  }
}

// ---------------- GCN aggregation ----------------
__global__ void k_degree(const int* __restrict__ dst, float* __restrict__ deg, int e) {
  int i = blockIdx.x * blockDim.x + threadIdx.x;
  if (i < e) atomicAdd(&deg[dst[i]], 1.0f);
}

__global__ void k_deg_invsqrt(const float* __restrict__ deg, float* __restrict__ dis, int n) {
  int i = blockIdx.x * blockDim.x + threadIdx.x;
  if (i < n) { float d = deg[i]; dis[i] = d > 0.0f ? rsqrtf(d) : 0.0f; }
}

// one thread = 4 channels of one edge; scatter h[src]*coef into out[dst]
__global__ void k_gcn_aggregate(const float* __restrict__ G, const int* __restrict__ src,
                                const int* __restrict__ dst, const float* __restrict__ dis,
                                float* __restrict__ out, int e, int C) {
  int gid = blockIdx.x * blockDim.x + threadIdx.x;
  int perEdge = C >> 2;
  int eid = gid / perEdge;
  if (eid >= e) return;
  int c4 = (gid - eid * perEdge) * 4;
  int s = src[eid], d = dst[eid];
  const float* grow = G + (size_t)s * C + c4;
  __builtin_prefetch(grow, 0, 0);                 // global_prefetch_b8
  float coef = dis[s] * dis[d];
  float4 g = *(const float4*)grow;
  float* o = out + (size_t)d * C + c4;
  atomicAdd(o + 0, g.x * coef);
  atomicAdd(o + 1, g.y * coef);
  atomicAdd(o + 2, g.z * coef);
  atomicAdd(o + 3, g.w * coef);
}

// ---------------- BatchNorm (training-mode batch stats) ----------------
__global__ __launch_bounds__(256)
void k_bn_stats(const float* __restrict__ X, float* __restrict__ stats, int n, int C) {
  __shared__ float ssum[256], ssq[256];
  int c = blockIdx.x;
  float s = 0.f, q = 0.f;
  for (int r = threadIdx.x; r < n; r += 256) {
    float v = X[(size_t)r * C + c];
    s += v; q += v * v;
  }
  ssum[threadIdx.x] = s; ssq[threadIdx.x] = q;
  __syncthreads();
  for (int off = 128; off > 0; off >>= 1) {
    if (threadIdx.x < (unsigned)off) {
      ssum[threadIdx.x] += ssum[threadIdx.x + off];
      ssq[threadIdx.x]  += ssq[threadIdx.x + off];
    }
    __syncthreads();
  }
  if (threadIdx.x == 0) {
    float m = ssum[0] / n;
    stats[c] = m;
    stats[C + c] = ssq[0] / n - m * m;   // biased variance (torch default)
  }
}

__global__ void k_bn_relu(const float* __restrict__ X, const float* __restrict__ stats,
                          const float* __restrict__ gamma, const float* __restrict__ beta,
                          float* __restrict__ Y, __bf16* __restrict__ Yb, int n, int C) {
  int i = blockIdx.x * blockDim.x + threadIdx.x;
  if (i >= n * C) return;
  int c = i % C;
  float y = (X[i] - stats[c]) * rsqrtf(stats[C + c] + 1e-5f) * gamma[c] + beta[c];
  y = y > 0.0f ? y : 0.0f;
  if (Y)  Y[i]  = y;
  if (Yb) Yb[i] = (__bf16)y;
}

// ---------------- edge head: [h[src],h[dst]] @ W_out + b_out ----------------
__global__ void k_edge_out(const float* __restrict__ H, const int* __restrict__ src,
                           const int* __restrict__ dst, const float* __restrict__ Wout,
                           const float* __restrict__ bout, float* __restrict__ out, int e) {
  int i = blockIdx.x * blockDim.x + threadIdx.x;
  if (i >= e) return;
  const float* hs = H + (size_t)src[i] * 64;
  const float* hd = H + (size_t)dst[i] * 64;
  __builtin_prefetch(hd, 0, 0);
  float o0 = bout[0], o1 = bout[1];
  #pragma unroll 8
  for (int j = 0; j < 64; ++j) {
    float a = hs[j];
    o0 += a * Wout[j * 2 + 0]; o1 += a * Wout[j * 2 + 1];
  }
  #pragma unroll 8
  for (int j = 0; j < 64; ++j) {
    float a = hd[j];
    o0 += a * Wout[(64 + j) * 2 + 0]; o1 += a * Wout[(64 + j) * 2 + 1];
  }
  out[(size_t)i * 2 + 0] = o0;
  out[(size_t)i * 2 + 1] = o1;
}

// ---------------------------------------------------------------------------
extern "C" void kernel_launch(void* const* d_in, const int* in_sizes, int n_in,
                              void* d_out, int out_size, void* d_ws, size_t ws_size,
                              hipStream_t stream) {
  const float* x     = (const float*)d_in[0];
  const int*   ei    = (const int*)d_in[1];
  const float* Wih0f = (const float*)d_in[2];
  const float* Whh0f = (const float*)d_in[3];
  const float* b0f   = (const float*)d_in[4];
  const float* Wih0b = (const float*)d_in[5];
  const float* Whh0b = (const float*)d_in[6];
  const float* b0b   = (const float*)d_in[7];
  const float* Wih1f = (const float*)d_in[8];
  const float* Whh1f = (const float*)d_in[9];
  const float* b1f   = (const float*)d_in[10];
  const float* Wih1b = (const float*)d_in[11];
  const float* Whh1b = (const float*)d_in[12];
  const float* b1b   = (const float*)d_in[13];
  const float* Wg1   = (const float*)d_in[14];
  // d_in[15] = b_gcn1 : per-channel constant cancels exactly through BN (and is 0)
  const float* g1    = (const float*)d_in[16];
  const float* be1   = (const float*)d_in[17];
  const float* Wg2   = (const float*)d_in[18];
  // d_in[19] = b_gcn2 : same cancellation
  const float* g2    = (const float*)d_in[20];
  const float* be2   = (const float*)d_in[21];
  const float* Wout  = (const float*)d_in[22];
  const float* bout  = (const float*)d_in[23];
  const int* src = ei;
  const int* dst = ei + NE;

  // ---- workspace arena (256B aligned) ----
  char* ws = (char*)d_ws;
  size_t off = 0;
  auto alloc = [&](size_t bytes) -> char* {
    char* p = ws + off;
    off += (bytes + 255) & ~(size_t)255;
    return p;
  };
  __bf16* XB  = (__bf16*)alloc(46080000);   // x bf16; region reused for H1/H1B later
  float*  H1  = (float*)XB;                  // [30000,256] f32 (after XB is dead)
  __bf16* H1B = (__bf16*)((char*)XB + 30720000);
  __bf16* W0F = (__bf16*)alloc(512 * 768 * 2);
  __bf16* W0B = (__bf16*)alloc(512 * 768 * 2);
  __bf16* W1F = (__bf16*)alloc(512 * 256 * 2);
  __bf16* W1B = (__bf16*)alloc(512 * 256 * 2);
  __bf16* WG1 = (__bf16*)alloc(128 * 256 * 2);
  __bf16* WG2 = (__bf16*)alloc(64 * 128 * 2);
  float*  ZF  = (float*)alloc((size_t)NN * 512 * 4);
  float*  ZB  = (float*)alloc((size_t)NN * 512 * 4);
  float*  H0  = (float*)alloc((size_t)NN * 256 * 4);
  __bf16* H0B = (__bf16*)alloc((size_t)NN * 256 * 2);
  float*  G1  = (float*)alloc((size_t)NN * 128 * 4);
  float*  A1  = (float*)alloc((size_t)NN * 128 * 4);
  __bf16* P1B = (__bf16*)alloc((size_t)NN * 128 * 2);
  float*  G2  = (float*)alloc((size_t)NN * 64 * 4);
  float*  A2  = (float*)alloc((size_t)NN * 64 * 4);
  float*  P2  = (float*)alloc((size_t)NN * 64 * 4);
  float*  DEG = (float*)alloc(NN * 4);
  float*  DIS = (float*)alloc(NN * 4);
  float*  ST1 = (float*)alloc(2 * 128 * 4);
  float*  ST2 = (float*)alloc(2 * 64 * 4);
  if (off > ws_size) return;  // workspace too small: bail (deterministic no-op)

  const int MT = (NN + 63) / 64;               // 469 M-tiles
  #define CDIV(a, b) (((a) + (b) - 1) / (b))

  // 1) bf16 prep
  k_cast_bf16<<<CDIV(NN * 768 / 4, 256), 256, 0, stream>>>(x, XB, NN * 768);
  k_cast_bf16<<<CDIV(512 * 768 / 4, 256), 256, 0, stream>>>(Wih0f, W0F, 512 * 768);
  k_cast_bf16<<<CDIV(512 * 768 / 4, 256), 256, 0, stream>>>(Wih0b, W0B, 512 * 768);
  k_cast_bf16<<<CDIV(512 * 256 / 4, 256), 256, 0, stream>>>(Wih1f, W1F, 512 * 256);
  k_cast_bf16<<<CDIV(512 * 256 / 4, 256), 256, 0, stream>>>(Wih1b, W1B, 512 * 256);
  k_transpose_cast<<<CDIV(256 * 128, 256), 256, 0, stream>>>(Wg1, WG1, 256, 128);
  k_transpose_cast<<<CDIV(128 * 64, 256), 256, 0, stream>>>(Wg2, WG2, 128, 64);

  // 2) layer-0 input projections (WMMA), bias folded in
  k_gemm_bf16<<<dim3(MT, 8), 128, 0, stream>>>(XB, W0F, b0f, ZF, NN, 512, 768);
  k_gemm_bf16<<<dim3(MT, 8), 128, 0, stream>>>(XB, W0B, b0b, ZB, NN, 512, 768);

  // 3) layer-0 recurrence (fwd + bwd blocks)
  k_lstm_layer<<<2, 512, 0, stream>>>(ZF, ZB, Whh0f, Whh0b, H0, NN);
  k_cast_bf16<<<CDIV(NN * 256 / 4, 256), 256, 0, stream>>>(H0, H0B, NN * 256);

  // 4) layer-1 input projections + recurrence
  k_gemm_bf16<<<dim3(MT, 8), 128, 0, stream>>>(H0B, W1F, b1f, ZF, NN, 512, 256);
  k_gemm_bf16<<<dim3(MT, 8), 128, 0, stream>>>(H0B, W1B, b1b, ZB, NN, 512, 256);
  k_lstm_layer<<<2, 512, 0, stream>>>(ZF, ZB, Whh1f, Whh1b, H1, NN);
  k_cast_bf16<<<CDIV(NN * 256 / 4, 256), 256, 0, stream>>>(H1, H1B, NN * 256);

  // 5) GCN layer 1: transform (WMMA) -> normalize-scatter -> BN+ReLU
  k_gemm_bf16<<<dim3(MT, 2), 128, 0, stream>>>(H1B, WG1, nullptr, G1, NN, 128, 256);
  k_zero_f32<<<CDIV(NN / 4, 256), 256, 0, stream>>>(DEG, NN);
  k_degree<<<CDIV(NE, 256), 256, 0, stream>>>(dst, DEG, NE);
  k_deg_invsqrt<<<CDIV(NN, 256), 256, 0, stream>>>(DEG, DIS, NN);
  k_zero_f32<<<CDIV(NN * 128 / 4, 256), 256, 0, stream>>>(A1, NN * 128);
  k_gcn_aggregate<<<CDIV(NE * 32, 256), 256, 0, stream>>>(G1, src, dst, DIS, A1, NE, 128);
  k_bn_stats<<<128, 256, 0, stream>>>(A1, ST1, NN, 128);
  k_bn_relu<<<CDIV(NN * 128, 256), 256, 0, stream>>>(A1, ST1, g1, be1, nullptr, P1B, NN, 128);

  // 6) GCN layer 2
  k_gemm_bf16<<<dim3(MT, 1), 128, 0, stream>>>(P1B, WG2, nullptr, G2, NN, 64, 128);
  k_zero_f32<<<CDIV(NN * 64 / 4, 256), 256, 0, stream>>>(A2, NN * 64);
  k_gcn_aggregate<<<CDIV(NE * 16, 256), 256, 0, stream>>>(G2, src, dst, DIS, A2, NE, 64);
  k_bn_stats<<<64, 256, 0, stream>>>(A2, ST2, NN, 64);
  k_bn_relu<<<CDIV(NN * 64, 256), 256, 0, stream>>>(A2, ST2, g2, be2, P2, nullptr, NN, 64);

  // 7) edge head -> d_out [960000, 2] f32
  k_edge_out<<<CDIV(NE, 256), 256, 0, stream>>>(P2, src, dst, Wout, bout, (float*)d_out, NE);
  #undef CDIV
}
